// CIN_13030930776368
// MI455X (gfx1250) — matrix operand order repaired and checked by reading, not compiled, not run on hardware
//
#include <hip/hip_runtime.h>

typedef __bf16 bf16_t;
typedef __attribute__((ext_vector_type(16))) __bf16 v16bf;
typedef __attribute__((ext_vector_type(8)))  __bf16 v8bf;
typedef __attribute__((ext_vector_type(8)))  float  v8f;

constexpr int kB   = 1024;
constexpr int kM   = 39;
constexpr int kD   = 64;
constexpr int kH1  = 128;
constexpr int kK0  = kM * kM;          // 1521
constexpr int kK1  = kM * kH1;         // 4992
constexpr int kNC0 = (kK0 + 31) / 32;  // 48 chunks (last one guarded)
constexpr int kNC1 = kK1 / 32;         // 156 chunks, exact
constexpr int XS_S = 68;               // fp32 row stride (bank-conflict pad)
constexpr int ZT_S = 40;               // bf16 row stride for Z^T (80B, conflict-free b128)
constexpr int ZT_BUF = kD * ZT_S;      // one Z^T chunk buffer (bf16 elems)

// exact floor(k/39) for 0 <= k <= 18724
__device__ __forceinline__ int div39(int k) {
  return (int)(((unsigned)k * 13444u) >> 19);
}

// One CIN layer: D = A(W) x B(Z) accumulated over K chunks of 32 (bf16 WMMA).
//   A[m=h, k] = W[k*128 + h0 + m]         (global, L2-resident)
//   B[k, n=d] = xs[i,d] * js[j,d]         (built on the fly into LDS, bf16)
// Producer (i,j) derived from scalar k via branchless magic arithmetic so all
// LDS loads of a chunk issue before a single wait.
template <bool L0>
__device__ __forceinline__ void cin_layer(
    const float* __restrict__ W,
    const float* __restrict__ xs,
    const float* __restrict__ js, int jstride,
    bf16_t* __restrict__ Zt,
    v8f acc[4], int tid)
{
  const int lane = tid & 31;
  const int m    = lane & 15;        // row within A tile / column within B tile
  const int hi16 = lane >> 4;        // lane half selector
  const int h0   = (tid >> 5) << 4;  // wave's h-tile base (0..112)
  const int d_p  = tid & 63;         // producer: d column (per-lane)
  // producer k sub-offset {0,8,16,24}: uniform per wave -> force scalar
  const int ks8  = __builtin_amdgcn_readfirstlane((tid >> 6) << 3);
  const int nch  = L0 ? kNC0 : kNC1;

#pragma unroll 1
  for (int c = 0; c < nch; ++c) {
    const int kbase = c << 5;
    bf16_t* zbuf = Zt + (c & 1) * ZT_BUF;   // double buffer

    // ---- cooperatively produce Z^T chunk: zbuf[d][kk] = bf16(x_i * y_j) ----
    {
      const int k0 = kbase + ks8;           // scalar
      float xv[8], yv[8];
#pragma unroll
      for (int e = 0; e < 8; ++e) {         // branchless: all 16 loads issue
        const int k = k0 + e;               // scalar
        int i, j;
        if (L0) { i = div39(k); j = k - i * kM; }   // scalar mul/shift/sub
        else    { i = k >> 7;   j = k & 127; }
        xv[e] = xs[i * XS_S + d_p];
        yv[e] = js[j * jstride + d_p];
      }
      union { v8bf v; bf16_t s[8]; } z;
      if (L0 && c == nch - 1) {             // guarded tail (K 1521 -> 1536 pad)
#pragma unroll
        for (int e = 0; e < 8; ++e) {
          const float p = xv[e] * yv[e];
          z.s[e] = (bf16_t)((k0 + e < kK0) ? p : 0.0f);
        }
      } else {
#pragma unroll
        for (int e = 0; e < 8; ++e)
          z.s[e] = (bf16_t)(xv[e] * yv[e]); // v_fma_mixlo/hi_bf16
      }
      *(v8bf*)(zbuf + d_p * ZT_S + ks8) = z.v;   // single 16B LDS store
    }
    __syncthreads();   // produce(c) visible; double buffering needs only this barrier

    // ---- A fragment (ISA 16-bit A layout: K(lane,e) = e + 8*(e>=8) + 8*(lane>=16)) ----
    union { v16bf v; bf16_t s[16]; } a;
    if (L0 && c == nch - 1) {           // guarded tail chunk
#pragma unroll
      for (int e = 0; e < 16; ++e) {
        const int k = kbase + (hi16 << 3) + e + ((e >= 8) ? 8 : 0);
        a.s[e] = (bf16_t)((k < kK0) ? W[k * 128 + h0 + m] : 0.0f);
      }
    } else {
      const float* pW = W + (kbase + (hi16 << 3)) * 128 + h0 + m;
      __builtin_prefetch(pW + 32 * 128, 0, 1);   // next chunk's A slice
#pragma unroll
      for (int e = 0; e < 16; ++e) {
        const int ko = e + ((e >= 8) ? 8 : 0);   // immediate-offset loads, stride 512B
        a.s[e] = (bf16_t)pW[ko * 128];
      }
    }

    // ---- gather all 4 B fragments, then 4 back-to-back WMMAs ----
    const int kk = hi16 << 3;           // B layout: elem e -> K = kk + e + 8*(e>=8), col = lane%16
    v16bf bv[4];
#pragma unroll
    for (int t = 0; t < 4; ++t) {
      const bf16_t* zp = zbuf + (t * 16 + m) * ZT_S + kk;
      v8bf lo = *(const v8bf*)(zp);
      v8bf hi = *(const v8bf*)(zp + 16);
      bv[t] = __builtin_shufflevector(lo, hi,
          0, 1, 2, 3, 4, 5, 6, 7, 8, 9, 10, 11, 12, 13, 14, 15);
    }
#pragma unroll
    for (int t = 0; t < 4; ++t)
      acc[t] = __builtin_amdgcn_wmma_f32_16x16x32_bf16(
          false, a.v, false, bv[t], (short)0, acc[t], false, false);
    // no trailing barrier: next iteration writes the other buffer
  }
}

__global__ __launch_bounds__(256, 1) void cin_wmma_kernel(
    const float* __restrict__ x,
    const float* __restrict__ W0,
    const float* __restrict__ W1,
    float* __restrict__ out)
{
  __shared__ __align__(16) float  xs [kM  * XS_S];
  __shared__ __align__(16) float  h1s[kH1 * XS_S];
  __shared__ __align__(16) bf16_t Zt [2 * ZT_BUF];

  const int b    = blockIdx.x;
  const int tid  = threadIdx.x;
  const int lane = tid & 31;
  const int m    = lane & 15;
  const int hi16 = lane >> 4;
  const int h0   = (tid >> 5) << 4;

  // stage x[b] (39x64 fp32) into padded LDS
  const float* xb = x + b * (kM * kD);
  for (int idx = tid; idx < kM * kD; idx += 256)
    xs[(idx >> 6) * XS_S + (idx & 63)] = xb[idx];
  __syncthreads();

  v8f acc[4] = {};

  // ---- layer 0: h1[h,d] = sum_k W0r[k,h] * (x_i x_j) ----
  cin_layer<true>(W0, xs, xs, XS_S, Zt, acc, tid);

  // C/D layout: VGPR r holds row m = r + 8*(lane>=16), col n = lane%16
#pragma unroll
  for (int t = 0; t < 4; ++t)
#pragma unroll
    for (int r = 0; r < 8; ++r)
      h1s[(h0 + r + 8 * hi16) * XS_S + t * 16 + m] = acc[t][r];

  {
    v8f tot = acc[0] + acc[1] + acc[2] + acc[3];   // fold the 4 d-tiles
#pragma unroll
    for (int r = 0; r < 8; ++r) {
      float s = tot[r];
      s += __shfl_xor(s, 1, 16);
      s += __shfl_xor(s, 2, 16);
      s += __shfl_xor(s, 4, 16);
      s += __shfl_xor(s, 8, 16);
      if (m == 0) out[b * 256 + h0 + r + 8 * hi16] = s;
    }
  }
  __syncthreads();   // h1s ready for layer-1 producers

  // ---- layer 1: h2[h,d] = sum_k W1r[k,h] * (x_i h1_j) ----
#pragma unroll
  for (int t = 0; t < 4; ++t) acc[t] = v8f{};
  cin_layer<false>(W1, xs, h1s, XS_S, Zt, acc, tid);

  {
    v8f tot = acc[0] + acc[1] + acc[2] + acc[3];
#pragma unroll
    for (int r = 0; r < 8; ++r) {
      float s = tot[r];
      s += __shfl_xor(s, 1, 16);
      s += __shfl_xor(s, 2, 16);
      s += __shfl_xor(s, 4, 16);
      s += __shfl_xor(s, 8, 16);
      if (m == 0) out[b * 256 + 128 + h0 + r + 8 * hi16] = s;
    }
  }
}

extern "C" void kernel_launch(void* const* d_in, const int* in_sizes, int n_in,
                              void* d_out, int out_size, void* d_ws, size_t ws_size,
                              hipStream_t stream) {
  (void)in_sizes; (void)n_in; (void)d_ws; (void)ws_size; (void)out_size;
  const float* x  = (const float*)d_in[0];
  const float* W0 = (const float*)d_in[1];
  const float* W1 = (const float*)d_in[2];
  float* out = (float*)d_out;
  cin_wmma_kernel<<<dim3(kB), dim3(256), 0, stream>>>(x, W0, W1, out);
}
